// ComplexGCN_14328010899645
// MI455X (gfx1250) — matrix operand order repaired and checked by reading, not compiled, not run on hardware
//
#include <hip/hip_runtime.h>

// Problem constants (from the reference)
#define NN    4096
#define EE    262144
#define DIN   256
#define HIDD  256
#define DOUT  128
#define HEADS 4
#define DH    64
#define LQ    4096

typedef _Float16 h16;
typedef __attribute__((ext_vector_type(16))) _Float16 v16h;
typedef __attribute__((ext_vector_type(8)))  _Float16 v8h;
typedef __attribute__((ext_vector_type(8)))  float    v8f;

#define WMMA(A, B, C) \
  __builtin_amdgcn_wmma_f32_16x16x32_f16(false, (A), false, (B), (short)0, (C), false, false)

__device__ __forceinline__ void atomAddF(float* p, float v) {
  __hip_atomic_fetch_add(p, v, __ATOMIC_RELAXED, __HIP_MEMORY_SCOPE_AGENT);
}

// ---------------- small utility kernels ----------------

__global__ void k_fill(float* __restrict__ p, float v, int n) {
  int i = blockIdx.x * blockDim.x + threadIdx.x;
  if (i < n) p[i] = v;
}

__global__ void k_deg(const int* __restrict__ col, const float* __restrict__ w,
                      float* __restrict__ deg, int e) {
  int i = blockIdx.x * blockDim.x + threadIdx.x;
  if (i < e) atomAddF(&deg[col[i]], w[i]);
}

__global__ void k_dinv(const float* __restrict__ deg, float* __restrict__ dinv, int n) {
  int i = blockIdx.x * blockDim.x + threadIdx.x;
  if (i < n) {
    float d = deg[i];
    dinv[i] = d > 0.f ? rsqrtf(d) : 0.f;
  }
}

// Wt[n*K + k] = (h16) W[k*Nc + n]   (transpose K x Nc -> Nc x K)
__global__ void k_cvt_t(const float* __restrict__ W, h16* __restrict__ Wt, int K, int Nc) {
  int i = blockIdx.x * blockDim.x + threadIdx.x;
  if (i >= K * Nc) return;
  int k = i / Nc, n = i % Nc;
  Wt[(size_t)n * K + k] = (h16)W[i];
}

// straight convert (already [out, in] row-major)
__global__ void k_cvt(const float* __restrict__ W, h16* __restrict__ Wt, int n) {
  int i = blockIdx.x * blockDim.x + threadIdx.x;
  if (i < n) Wt[i] = (h16)W[i];
}

// ---------------- WMMA GEMM: C[M,Nc] = A[M,K](f32) * Bt[Nc,K](f16)^T (+bias, act) ----------------
// act: 0 = none, 1 = relu
__global__ void k_gemm(const float* __restrict__ A, const h16* __restrict__ Bt,
                       float* __restrict__ C, const float* __restrict__ bias,
                       int M, int Nc, int K, int act) {
  const int lane = threadIdx.x & 31;
  const int wave = threadIdx.x >> 5;
  const int tiles_m = M >> 4;
  const int tiles_n = Nc >> 6;
  int task = blockIdx.x * 8 + wave;
  if (task >= tiles_m * tiles_n) return;
  const int tm = task % tiles_m;
  const int tn = task / tiles_m;
  const int hi = lane >> 4, nn = lane & 15;

  v8f acc[4];
  const v8f zf = {0.f, 0.f, 0.f, 0.f, 0.f, 0.f, 0.f, 0.f};
#pragma unroll
  for (int t = 0; t < 4; ++t) acc[t] = zf;

  const float* arow = A + (size_t)(tm * 16 + nn) * K;
  for (int k0 = 0; k0 < K; k0 += 32) {
    // A operand: lane = row m (nn); halves j = K k0+16*hi+j
    v16h a;
    const float* ap = arow + k0 + 16 * hi;
#pragma unroll
    for (int j = 0; j < 8; ++j) {
      a[j]     = (h16)ap[j];
      a[8 + j] = (h16)ap[8 + j];
    }
#pragma unroll
    for (int t = 0; t < 4; ++t) {
      // B operand: lane = col n; halves j = K k0+16*hi+j, contiguous in Bt row
      const h16* bp = Bt + (size_t)(tn * 64 + t * 16 + nn) * K + k0 + 16 * hi;
      v16h b = *(const v16h*)bp;
      acc[t] = WMMA(a, b, acc[t]);
    }
  }

#pragma unroll
  for (int t = 0; t < 4; ++t) {
    int colg = tn * 64 + t * 16 + nn;
    float bv = bias ? bias[colg] : 0.f;
#pragma unroll
    for (int r = 0; r < 8; ++r) {
      int rowg = tm * 16 + r + 8 * hi;
      float v = acc[t][r] + bv;
      if (act == 1) v = fmaxf(v, 0.f);
      C[(size_t)rowg * Nc + colg] = v;
    }
  }
}

// ---------------- GCN scatter (normalized aggregation) ----------------

// x[i,:] = dinv[i]^2 * h[i,:] + bias   (self-loop + bias init)
__global__ void k_scatter_init(const float* __restrict__ h, const float* __restrict__ dinv,
                               const float* __restrict__ bias, float* __restrict__ x,
                               int n, int C, int lgC) {
  int idx = blockIdx.x * blockDim.x + threadIdx.x;
  int i4 = idx * 4;
  if (i4 >= n * C) return;
  int node = i4 >> lgC;
  int c = i4 & (C - 1);
  float d = dinv[node];
  float dd = d * d;
  float4 hv = *(const float4*)(h + (size_t)node * C + c);
  float4 bv = *(const float4*)(bias + c);
  float4 r;
  r.x = hv.x * dd + bv.x;
  r.y = hv.y * dd + bv.y;
  r.z = hv.z * dd + bv.z;
  r.w = hv.w * dd + bv.w;
  *(float4*)(x + (size_t)node * C + c) = r;
}

// x[col,:] += dinv[row]*w*dinv[col] * h[row,:]   — one thread per (edge, 4-feature chunk)
__global__ void k_scatter_edges(const int* __restrict__ row, const int* __restrict__ col,
                                const float* __restrict__ w, const float* __restrict__ dinv,
                                const float* __restrict__ h, float* __restrict__ x,
                                int e, int C, int lgcpt) {
  int idx = blockIdx.x * blockDim.x + threadIdx.x;
  int ed = idx >> lgcpt;
  if (ed >= e) return;
  int c4 = (idx & ((1 << lgcpt) - 1)) * 4;
  int r = row[ed], cl = col[ed];
  float nw = dinv[r] * w[ed] * dinv[cl];
  float4 hv = *(const float4*)(h + (size_t)r * C + c4);
  float* xp = x + (size_t)cl * C + c4;
  atomAddF(xp + 0, nw * hv.x);
  atomAddF(xp + 1, nw * hv.y);
  atomAddF(xp + 2, nw * hv.z);
  atomAddF(xp + 3, nw * hv.w);
}

// ---------------- MHA ----------------

// qkv f32 [L, 768] -> Qh f16 [H,L,64] (scaled by 1/8, exact pow2), Kh f16 [H,L,64], Vt f16 [H,64,L]
__global__ void k_split_qkv(const float* __restrict__ qkv, h16* __restrict__ Qh,
                            h16* __restrict__ Kh, h16* __restrict__ Vt) {
  int i = blockIdx.x * blockDim.x + threadIdx.x;
  if (i >= LQ * 3 * HIDD) return;
  int l = i / (3 * HIDD);
  int c = i % (3 * HIDD);
  float v = qkv[i];
  int part = c >> 8;          // 0=q, 1=k, 2=v
  int cc = c & 255;
  int head = cc >> 6;
  int d = cc & 63;
  if (part == 0)
    Qh[((size_t)head * LQ + l) * DH + d] = (h16)(v * 0.125f);  // fold 1/sqrt(64)
  else if (part == 1)
    Kh[((size_t)head * LQ + l) * DH + d] = (h16)v;
  else
    Vt[((size_t)head * DH + d) * LQ + l] = (h16)v;
}

// Two-pass flash attention: each wave owns (head, 16-query-row block). O is [L, 256] f32.
__global__ void k_attn(const h16* __restrict__ Qh, const h16* __restrict__ Kh,
                       const h16* __restrict__ Vt, float* __restrict__ O) {
  __shared__ h16 ptile[8 * 512];  // 8 waves x (16 x 32) f16 P tiles
  const int lane = threadIdx.x & 31;
  const int wave = threadIdx.x >> 5;
  const int task = blockIdx.x * 8 + wave;
  if (task >= HEADS * (LQ / 16)) return;
  const int head = task >> 8;   // LQ/16 == 256
  const int qb = task & 255;
  const int hi = lane >> 4, nn = lane & 15;

  // Q A-operands for the two K=32 steps of dh=64
  v16h qa[2];
  const h16* qrow = Qh + ((size_t)head * LQ + qb * 16 + nn) * DH;
#pragma unroll
  for (int s = 0; s < 2; ++s) {
    v8h t0 = *(const v8h*)(qrow + s * 32 + 16 * hi);
    v8h t1 = *(const v8h*)(qrow + s * 32 + 16 * hi + 8);
#pragma unroll
    for (int j = 0; j < 8; ++j) { qa[s][j] = t0[j]; qa[s][8 + j] = t1[j]; }
  }

  float m[8], l[8];
#pragma unroll
  for (int r = 0; r < 8; ++r) { m[r] = -1e30f; l[r] = 0.f; }

  // -------- pass 1: row max + sum of exp --------
  for (int j0 = 0; j0 < LQ; j0 += 16) {
    const h16* krow = Kh + ((size_t)head * LQ + j0 + nn) * DH;
    v16h b0 = *(const v16h*)(krow + 16 * hi);
    v16h b1 = *(const v16h*)(krow + 32 + 16 * hi);
    v8f s = {0.f, 0.f, 0.f, 0.f, 0.f, 0.f, 0.f, 0.f};
    s = WMMA(qa[0], b0, s);
    s = WMMA(qa[1], b1, s);
#pragma unroll
    for (int r = 0; r < 8; ++r) {
      float v = s[r];
      float mx = v;
      mx = fmaxf(mx, __shfl_xor(mx, 1, 32));
      mx = fmaxf(mx, __shfl_xor(mx, 2, 32));
      mx = fmaxf(mx, __shfl_xor(mx, 4, 32));
      mx = fmaxf(mx, __shfl_xor(mx, 8, 32));
      float mn = fmaxf(m[r], mx);
      float e = __expf(v - mn);
      e += __shfl_xor(e, 1, 32);
      e += __shfl_xor(e, 2, 32);
      e += __shfl_xor(e, 4, 32);
      e += __shfl_xor(e, 8, 32);
      l[r] = l[r] * __expf(m[r] - mn) + e;
      m[r] = mn;
    }
  }

  // -------- pass 2: O = sum exp(s - m) * V --------
  v8f o[4];
  const v8f zf = {0.f, 0.f, 0.f, 0.f, 0.f, 0.f, 0.f, 0.f};
#pragma unroll
  for (int t = 0; t < 4; ++t) o[t] = zf;

  h16* myP = ptile + wave * 512;
  for (int j0 = 0; j0 < LQ; j0 += 32) {
#pragma unroll
    for (int t = 0; t < 2; ++t) {
      const h16* krow = Kh + ((size_t)head * LQ + j0 + t * 16 + nn) * DH;
      v16h b0 = *(const v16h*)(krow + 16 * hi);
      v16h b1 = *(const v16h*)(krow + 32 + 16 * hi);
      v8f s = zf;
      s = WMMA(qa[0], b0, s);
      s = WMMA(qa[1], b1, s);
#pragma unroll
      for (int r = 0; r < 8; ++r) {
        float p = __expf(s[r] - m[r]);
        myP[(r + 8 * hi) * 32 + t * 16 + nn] = (h16)p;   // C-layout -> row-major LDS
      }
    }
    // reload P tile in A-operand layout (wave-private; DS waits inserted by compiler)
    v16h pa;
    {
      v8h t0 = *(const v8h*)(myP + nn * 32 + 16 * hi);
      v8h t1 = *(const v8h*)(myP + nn * 32 + 16 * hi + 8);
#pragma unroll
      for (int j = 0; j < 8; ++j) { pa[j] = t0[j]; pa[8 + j] = t1[j]; }
    }
#pragma unroll
    for (int t = 0; t < 4; ++t) {
      const h16* vrow = Vt + ((size_t)head * DH + t * 16 + nn) * LQ + j0 + 16 * hi;
      v16h bv = *(const v16h*)vrow;
      o[t] = WMMA(pa, bv, o[t]);
    }
  }

  float linv[8];
#pragma unroll
  for (int r = 0; r < 8; ++r) linv[r] = 1.0f / l[r];
#pragma unroll
  for (int t = 0; t < 4; ++t) {
#pragma unroll
    for (int r = 0; r < 8; ++r) {
      int rowg = qb * 16 + r + 8 * hi;
      int colg = head * DH + t * 16 + nn;
      O[(size_t)rowg * HIDD + colg] = o[t][r] * linv[r];
    }
  }
}

__global__ void k_sigmoid(const float* __restrict__ x, float* __restrict__ out, int n) {
  int i = blockIdx.x * blockDim.x + threadIdx.x;
  if (i < n) out[i] = 1.0f / (1.0f + __expf(-x[i]));
}

// ---------------- driver ----------------

extern "C" void kernel_launch(void* const* d_in, const int* in_sizes, int n_in,
                              void* d_out, int out_size, void* d_ws, size_t ws_size,
                              hipStream_t stream) {
  (void)in_sizes; (void)n_in; (void)out_size; (void)ws_size;
  const float* x0   = (const float*)d_in[0];
  const int*   ei   = (const int*)d_in[1];
  const float* ew   = (const float*)d_in[2];
  const float* W1   = (const float*)d_in[3];
  const float* b1   = (const float*)d_in[4];
  const float* W2   = (const float*)d_in[5];
  const float* b2   = (const float*)d_in[6];
  const float* W3   = (const float*)d_in[7];
  const float* b3   = (const float*)d_in[8];
  const float* inW  = (const float*)d_in[9];
  const float* inB  = (const float*)d_in[10];
  const float* outW = (const float*)d_in[11];
  const float* outB = (const float*)d_in[12];
  float* out = (float*)d_out;

  const int* row = ei;
  const int* col = ei + EE;

  char* base = (char*)d_ws;
  size_t off = 0;
  auto alloc = [&](size_t bytes) -> void* {
    off = (off + 255) & ~(size_t)255;
    void* p = base + off;
    off += bytes;
    return p;
  };
  float* deg   = (float*)alloc((size_t)NN * 4);
  float* dinv  = (float*)alloc((size_t)NN * 4);
  h16*   W1t   = (h16*)alloc((size_t)DIN * HIDD * 2);
  h16*   W2t   = (h16*)alloc((size_t)HIDD * HIDD * 2);
  h16*   W3t   = (h16*)alloc((size_t)HIDD * DOUT * 2);
  h16*   inPt  = (h16*)alloc((size_t)3 * HIDD * HIDD * 2);
  h16*   outPt = (h16*)alloc((size_t)HIDD * HIDD * 2);
  float* hbuf  = (float*)alloc((size_t)NN * HIDD * 4);
  float* xA    = (float*)alloc((size_t)NN * HIDD * 4);
  float* xB    = (float*)alloc((size_t)NN * HIDD * 4);
  float* qkv   = (float*)alloc((size_t)NN * 3 * HIDD * 4);
  h16*   Qh    = (h16*)alloc((size_t)HEADS * LQ * DH * 2);
  h16*   Kh    = (h16*)alloc((size_t)HEADS * LQ * DH * 2);
  h16*   Vt    = (h16*)alloc((size_t)HEADS * DH * LQ * 2);
  float* attnO = (float*)alloc((size_t)NN * HIDD * 4);

  // --- GCN normalization terms (shared by all layers) ---
  k_fill<<<NN / 256, 256, 0, stream>>>(deg, 1.0f, NN);                 // self-loop weight
  k_deg<<<EE / 256, 256, 0, stream>>>(col, ew, deg, EE);
  k_dinv<<<NN / 256, 256, 0, stream>>>(deg, dinv, NN);

  // --- weights -> f16, B-operand (N-major) layout ---
  k_cvt_t<<<(DIN * HIDD) / 256, 256, 0, stream>>>(W1, W1t, DIN, HIDD);
  k_cvt_t<<<(HIDD * HIDD) / 256, 256, 0, stream>>>(W2, W2t, HIDD, HIDD);
  k_cvt_t<<<(HIDD * DOUT) / 256, 256, 0, stream>>>(W3, W3t, HIDD, DOUT);
  k_cvt<<<(3 * HIDD * HIDD) / 256, 256, 0, stream>>>(inW, inPt, 3 * HIDD * HIDD);
  k_cvt<<<(HIDD * HIDD) / 256, 256, 0, stream>>>(outW, outPt, HIDD * HIDD);

  auto gemm = [&](const float* A, const h16* Bt, float* C, const float* bias,
                  int M, int Nc, int K, int act) {
    int waves = (M >> 4) * (Nc >> 6);
    k_gemm<<<waves / 8, 256, 0, stream>>>(A, Bt, C, bias, M, Nc, K, act);
  };
  auto gcn = [&](const float* xin, const h16* Wt, const float* bias, float* xout,
                 int C, int lgC, int lgcpt) {
    gemm(xin, Wt, hbuf, nullptr, NN, C, 256, 0);
    k_scatter_init<<<(NN * C / 4) / 256, 256, 0, stream>>>(hbuf, dinv, bias, xout, NN, C, lgC);
    k_scatter_edges<<<(EE * (C / 4)) / 256, 256, 0, stream>>>(row, col, ew, dinv, hbuf, xout,
                                                             EE, C, lgcpt);
  };
  auto mhaRelu = [&](const float* xin, float* xout) {
    gemm(xin, inPt, qkv, inB, NN, 3 * HIDD, 256, 0);
    k_split_qkv<<<(NN * 3 * HIDD) / 256, 256, 0, stream>>>(qkv, Qh, Kh, Vt);
    k_attn<<<(HEADS * (LQ / 16)) / 8, 256, 0, stream>>>(Qh, Kh, Vt, attnO);
    gemm(attnO, outPt, xout, outB, NN, HIDD, 256, 1);  // relu fused
  };

  gcn(x0, W1t, b1, xA, 256, 8, 6);
  mhaRelu(xA, xB);
  gcn(xB, W2t, b2, xA, 256, 8, 6);
  mhaRelu(xA, xB);
  gcn(xB, W3t, b3, xA, 128, 7, 5);
  k_sigmoid<<<(NN * DOUT) / 256, 256, 0, stream>>>(xA, out, NN * DOUT);
}
